// SPELL_86079734546822
// MI455X (gfx1250) — matrix operand order repaired and checked by reading, not compiled, not run on hardware
//
#include <hip/hip_runtime.h>
#include <stdint.h>

typedef __attribute__((ext_vector_type(16))) __bf16 v16bf;
typedef __attribute__((ext_vector_type(8)))  float  v8f;
typedef __attribute__((ext_vector_type(16))) unsigned short v16us;

// ---------- helpers ----------
__device__ __forceinline__ unsigned short f2bf(float f) {
  unsigned u = __float_as_uint(f);
  u += 0x7FFFu + ((u >> 16) & 1u);           // round-to-nearest-even
  return (unsigned short)(u >> 16);
}
// monotone float -> uint key (for atomic max on floats)
__device__ __forceinline__ unsigned fkey(float f) {
  unsigned u = __float_as_uint(f);
  return (u & 0x80000000u) ? ~u : (u | 0x80000000u);
}
__device__ __forceinline__ float keyf(unsigned k) {
  unsigned u = (k & 0x80000000u) ? (k & 0x7FFFFFFFu) : ~k;
  return __uint_as_float(u);
}
#define NEGINF_KEY 0x007FFFFFu   // fkey(-inf)

// A fragment: 16x32 bf16 tile, row-major in LDS (ushort), stride ldz.
// lane<16: row=lane, K = k0+0..7 and k0+16..23 ; lane>=16: row=lane-16, K = k0+8..15, k0+24..31
__device__ __forceinline__ v16bf load_a_frag(const unsigned short* z, int ldz, int k0) {
  int lane = threadIdx.x & 31;
  int row  = lane & 15;
  int koff = k0 + ((lane & 16) ? 8 : 0);
  const unsigned short* p = z + row * ldz + koff;
  v16us t;
#pragma unroll
  for (int j = 0; j < 8; ++j) { t[j] = p[j]; t[8 + j] = p[16 + j]; }
  return __builtin_bit_cast(v16bf, t);
}
// B fragment from pre-packed buffer: frag-major, 32 lanes x 16 bf16 contiguous per lane
__device__ __forceinline__ v16bf load_b_frag(const unsigned short* pack, int frag) {
  int lane = threadIdx.x & 31;
  const unsigned short* p = pack + ((size_t)frag * 32 + lane) * 16;
  v16us t;
#pragma unroll
  for (int j = 0; j < 16; ++j) t[j] = p[j];
  return __builtin_bit_cast(v16bf, t);
}
__device__ __forceinline__ v8f wmma_bf16(v16bf a, v16bf b, v8f c) {
  return __builtin_amdgcn_wmma_f32_16x16x32_bf16(false, a, false, b, (short)0, c, false, false);
}

// ---------- utility kernels ----------
__global__ void k_zero_f32(float* p, int n) {
  for (int i = blockIdx.x * blockDim.x + threadIdx.x; i < n; i += gridDim.x * blockDim.x) p[i] = 0.f;
}
__global__ void k_fill_u32(unsigned* p, int n, unsigned v) {
  for (int i = blockIdx.x * blockDim.x + threadIdx.x; i < n; i += gridDim.x * blockDim.x) p[i] = v;
}

// Pack row-major f32 weight [K x Nout] (rows < Ksplit from Wa, else Wb[k-Ksplit]) into
// bf16 WMMA-B fragment order: frag = ks*(Nout/16)+nt, per-lane 16 K values: ks*32 + 8h+0..7, 16+8h+0..7 (h = lane/16).
__global__ void k_pack_b(const float* __restrict__ Wa, const float* __restrict__ Wb,
                         int Ksplit, int K, int Nout, unsigned short* __restrict__ out) {
  int total = (K / 32) * (Nout / 16) * 32;
  int i = blockIdx.x * blockDim.x + threadIdx.x;
  if (i >= total) return;
  int lane = i & 31, frag = i >> 5;
  int ntiles = Nout >> 4;
  int ks = frag / ntiles, nt = frag % ntiles;
  int col = nt * 16 + (lane & 15);
  int kbase = ks * 32 + ((lane & 16) ? 8 : 0);
  unsigned short* o = out + (size_t)i * 16;
#pragma unroll
  for (int j = 0; j < 16; ++j) {
    int k = kbase + (j < 8 ? j : j + 8);
    float v = (k < Ksplit) ? Wa[(size_t)k * Nout + col] : Wb[(size_t)(k - Ksplit) * Nout + col];
    o[j] = f2bf(v);
  }
}

// Fold spf path: W4[4x64] = spf_w @ l011_w[512:576], bfold = spf_b @ l011_w[512:576] + l011_b + l012_b
__global__ void k_fold_spf(const float* spf_w, const float* spf_b, const float* l011_w,
                           const float* l011_b, const float* l012_b, float* W4, float* bfold) {
  int c = threadIdx.x;  // 64 threads
  float acc[4] = {0.f, 0.f, 0.f, 0.f};
  float bb = 0.f;
  for (int h = 0; h < 64; ++h) {
    float wcol = l011_w[(size_t)(512 + h) * 64 + c];
#pragma unroll
    for (int q = 0; q < 4; ++q) acc[q] += spf_w[q * 64 + h] * wcol;
    bb += spf_b[h] * wcol;
  }
#pragma unroll
  for (int q = 0; q < 4; ++q) W4[q * 64 + c] = acc[q];
  bfold[c] = bb + l011_b[c] + l012_b[c];
}

// ---------- node projection GEMM (WMMA): pre = x[:, :1024] @ Wcat + spf@W4 + bfold ----------
__global__ void k_node_gemm(const float* __restrict__ x, const unsigned short* __restrict__ wcat,
                            const float* __restrict__ W4, const float* __restrict__ bfold,
                            float* __restrict__ pre, int nrows) {
  __shared__ unsigned short a_lds[16 * 32];
  int tid = threadIdx.x;          // 32 threads = 1 wave
  int r0 = blockIdx.x * 16;
  v8f acc[4];
#pragma unroll
  for (int nt = 0; nt < 4; ++nt) acc[nt] = (v8f){0.f,0.f,0.f,0.f,0.f,0.f,0.f,0.f};

  int row = tid & 15, half = tid >> 4;
  int gr = r0 + row;
  for (int k0 = 0; k0 < 1024; k0 += 32) {
    unsigned short* dl = a_lds + row * 32 + half * 16;
    if (gr < nrows) {
      const float* xp = x + (size_t)gr * 1028 + k0 + half * 16;
#pragma unroll
      for (int j = 0; j < 16; ++j) dl[j] = f2bf(xp[j]);
    } else {
#pragma unroll
      for (int j = 0; j < 16; ++j) dl[j] = 0;
    }
    __syncthreads();
    v16bf a = load_a_frag(a_lds, 32, 0);
#pragma unroll
    for (int nt = 0; nt < 4; ++nt) {
      v16bf b = load_b_frag(wcat, (k0 >> 5) * 4 + nt);
      acc[nt] = wmma_bf16(a, b, acc[nt]);
    }
    __syncthreads();
  }
  // epilogue: add folded spf path + bias, store f32
  int n0 = tid & 15, moff = (tid & 16) ? 8 : 0;
#pragma unroll
  for (int nt = 0; nt < 4; ++nt) {
    int col = nt * 16 + n0;
#pragma unroll
    for (int j = 0; j < 8; ++j) {
      int r = r0 + j + moff;
      if (r < nrows) {
        const float* spf = x + (size_t)r * 1028 + 1024;
        float v = acc[nt][j] + bfold[col];
#pragma unroll
        for (int q = 0; q < 4; ++q) v += spf[q] * W4[q * 64 + col];
        pre[(size_t)r * 64 + col] = v;
      }
    }
  }
}

// ---------- BN ----------
__global__ void k_bn_stats(const float* __restrict__ h, int n_elems, float* __restrict__ stats) {
  __shared__ float ls[128];
  int t = threadIdx.x;
  if (t < 128) ls[t] = 0.f;
  __syncthreads();
  for (int i = blockIdx.x * blockDim.x + t; i < n_elems; i += gridDim.x * blockDim.x) {
    float v = h[i]; int c = i & 63;
    atomicAdd(&ls[c], v);
    atomicAdd(&ls[64 + c], v * v);
  }
  __syncthreads();
  if (t < 128) atomicAdd(&stats[t], ls[t]);
}
__global__ void k_bn_relu(float* __restrict__ h, int n_elems, const float* __restrict__ stats,
                          const float* __restrict__ g, const float* __restrict__ b, float invN) {
  int i = blockIdx.x * blockDim.x + threadIdx.x;
  if (i >= n_elems) return;
  int c = i & 63;
  float m = stats[c] * invN;
  float var = stats[64 + c] * invN - m * m;
  float y = g[c] * (h[i] - m) * rsqrtf(var + 1e-5f) + b[c];
  h[i] = y > 0.f ? y : 0.f;
}

// ---------- edge MLP (WMMA) + masked segment-max scatter ----------
// 1 wave per block, 64 edges per wave (4 M-tiles); one B-fragment load feeds 4 WMMAs.
#define ET 4              // M-tiles per wave
#define EPB (ET * 16)     // edges per block
__global__ void k_edge_mlp(const float* __restrict__ h0, const int* __restrict__ src,
                           const int* __restrict__ dst, const float* __restrict__ attr,
                           const unsigned short* __restrict__ w1p, const float* __restrict__ b1,
                           const unsigned short* __restrict__ w2p, const float* __restrict__ b2,
                           int hid, int nedges,
                           unsigned* acc1, unsigned* acc2, unsigned* accA) {
  __shared__ unsigned short z[EPB * 128];    // [edge][128] = [h_dst, h_src-h_dst]  (16KB)
  __shared__ unsigned short hb[EPB * 256];   // hidden activations, row stride 256  (32KB)
  int tid = threadIdx.x;                     // 32 threads = 1 wave
  int e0 = blockIdx.x * EPB;
  // gather: tasks = (edge, half) pairs; 2*EPB tasks over 32 lanes
  for (int task = tid; task < 2 * EPB; task += 32) {
    int er = task >> 1, half = task & 1;
    int e = e0 + er;
    unsigned short* zr = z + er * 128;
    if (e < nedges) {
      int s = src[e], d = dst[e];
      const float* hd = h0 + (size_t)d * 64 + half * 32;
      const float* hs = h0 + (size_t)s * 64 + half * 32;
#pragma unroll
      for (int j = 0; j < 32; ++j) {
        float a = hd[j], bq = hs[j];
        zr[half * 32 + j] = f2bf(a);
        zr[64 + half * 32 + j] = f2bf(bq - a);
      }
    } else {
#pragma unroll
      for (int j = 0; j < 32; ++j) { zr[half * 32 + j] = 0; zr[64 + half * 32 + j] = 0; }
    }
  }
  __syncthreads();
  int n0 = tid & 15, moff = (tid & 16) ? 8 : 0;
  // layer 1: z[EPB x 128] @ w1 -> relu -> hb[EPB x hid]
  int nt1 = hid >> 4;
  for (int nt = 0; nt < nt1; ++nt) {
    v8f c[ET];
#pragma unroll
    for (int mt = 0; mt < ET; ++mt) c[mt] = (v8f){0.f,0.f,0.f,0.f,0.f,0.f,0.f,0.f};
#pragma unroll
    for (int ks = 0; ks < 4; ++ks) {
      int frag = ks * nt1 + nt;
      __builtin_prefetch(w1p + ((size_t)(frag + 1) * 32 + (tid & 31)) * 16, 0, 1);
      v16bf b = load_b_frag(w1p, frag);
#pragma unroll
      for (int mt = 0; mt < ET; ++mt) {
        v16bf a = load_a_frag(z + mt * 16 * 128, 128, ks * 32);
        c[mt] = wmma_bf16(a, b, c[mt]);
      }
    }
    float bias = b1[nt * 16 + n0];
#pragma unroll
    for (int mt = 0; mt < ET; ++mt) {
#pragma unroll
      for (int j = 0; j < 8; ++j) {
        float v = c[mt][j] + bias;
        hb[(mt * 16 + j + moff) * 256 + nt * 16 + n0] = f2bf(v > 0.f ? v : 0.f);
      }
    }
  }
  __syncthreads();
  // layer 2: hb[EPB x hid] @ w2 -> m[EPB x 64], scatter atomic-max into masked accumulators
  int ks2 = hid >> 5;
  for (int nt = 0; nt < 4; ++nt) {
    v8f c[ET];
#pragma unroll
    for (int mt = 0; mt < ET; ++mt) c[mt] = (v8f){0.f,0.f,0.f,0.f,0.f,0.f,0.f,0.f};
    for (int ks = 0; ks < ks2; ++ks) {
      int frag = ks * 4 + nt;
      __builtin_prefetch(w2p + ((size_t)(frag + 1) * 32 + (tid & 31)) * 16, 0, 1);
      v16bf b = load_b_frag(w2p, frag);
#pragma unroll
      for (int mt = 0; mt < ET; ++mt) {
        v16bf a = load_a_frag(hb + mt * 16 * 256, 256, ks * 32);
        c[mt] = wmma_bf16(a, b, c[mt]);
      }
    }
    float bias = b2[nt * 16 + n0];
#pragma unroll
    for (int mt = 0; mt < ET; ++mt) {
#pragma unroll
      for (int j = 0; j < 8; ++j) {
        int e = e0 + mt * 16 + j + moff;
        if (e < nedges) {
          float v = c[mt][j] + bias;
          unsigned key = fkey(v);
          size_t idx = (size_t)dst[e] * 64 + nt * 16 + n0;
          float a = attr[e];
          if (accA) atomicMax(accA + idx, key);
          if (acc1 && a >= 0.f) atomicMax(acc1 + idx, key);
          if (acc2 && a <= 0.f) atomicMax(acc2 + idx, key);
        }
      }
    }
  }
}

__global__ void k_seg_finalize(const unsigned* __restrict__ acc, float* __restrict__ h, int n) {
  for (int i = blockIdx.x * blockDim.x + threadIdx.x; i < n; i += gridDim.x * blockDim.x) {
    float v = keyf(acc[i]);
    unsigned u = __float_as_uint(v);
    if ((u & 0x7F800000u) == 0x7F800000u) v = 0.f;  // non-finite -> 0
    h[i] = v;
  }
}

// ---------- SAGE ----------
__global__ void k_counts(const int* __restrict__ dst, const float* __restrict__ attr, int E,
                         float* c1, float* c2, float* cA) {
  int e = blockIdx.x * blockDim.x + threadIdx.x;
  if (e >= E) return;
  int d = dst[e]; float a = attr[e];
  if (a >= 0.f) atomicAdd(&c1[d], 1.f);
  if (a <= 0.f) atomicAdd(&c2[d], 1.f);
  atomicAdd(&cA[d], 1.f);
}
__global__ void k_sage_scatter(const float* __restrict__ h, const int* __restrict__ src,
                               const int* __restrict__ dst, const float* __restrict__ attr,
                               int maskmode, int E, float* __restrict__ sums) {
  int total = E * 64;
  for (int i = blockIdx.x * blockDim.x + threadIdx.x; i < total; i += gridDim.x * blockDim.x) {
    int e = i >> 6, c = i & 63;
    float a = attr[e];
    bool ok = (maskmode == 2) || (maskmode == 0 ? (a >= 0.f) : (a <= 0.f));
    if (ok) atomicAdd(&sums[(size_t)dst[e] * 64 + c], h[(size_t)src[e] * 64 + c]);
  }
}
__global__ void k_sage_linear(const float* __restrict__ h, const float* __restrict__ sums,
                              const float* __restrict__ cnt, const float* __restrict__ wl,
                              const float* __restrict__ bl, const float* __restrict__ wr,
                              float* __restrict__ out, int nrows) {
  __shared__ float hr[64], mr[64];
  int n = blockIdx.x, t = threadIdx.x;
  if (n >= nrows) return;
  float inv = 1.f / fmaxf(cnt[n], 1.f);
  hr[t] = h[(size_t)n * 64 + t];
  mr[t] = sums[(size_t)n * 64 + t] * inv;
  __syncthreads();
  float acc = bl[t];
  for (int k = 0; k < 64; ++k) acc += mr[k] * wl[k * 64 + t] + hr[k] * wr[k * 64 + t];
  out[(size_t)n * 64 + t] = acc;
}

// ---------- heads ----------
__global__ void k_head_dots(const float* __restrict__ xb, const float* __restrict__ wl,
                            const float* __restrict__ wr, float* __restrict__ t_out,
                            float* __restrict__ u_out, int nrows) {
  int n = blockIdx.x * blockDim.x + threadIdx.x;
  if (n >= nrows) return;
  const float* h = xb + (size_t)n * 64;
  float a = 0.f, b = 0.f;
#pragma unroll 4
  for (int k = 0; k < 64; ++k) { a += h[k] * wl[k]; b += h[k] * wr[k]; }
  t_out[n] = a; u_out[n] = b;
}
__global__ void k_head_scatter(const float* __restrict__ t_all, const int* __restrict__ src,
                               const int* __restrict__ dst, const float* __restrict__ attr,
                               float* __restrict__ hacc, int E, int N) {
  int e = blockIdx.x * blockDim.x + threadIdx.x;
  if (e >= E) return;
  int s = src[e], d = dst[e];
  float a = attr[e];
  bool m1 = a >= 0.f, m2 = a <= 0.f;
#pragma unroll
  for (int i = 0; i < 9; ++i) {
    int g = i % 3;
    bool ok = (g == 0) ? m1 : (g == 1) ? m2 : true;
    if (ok) atomicAdd(&hacc[(size_t)i * N + d], t_all[(size_t)i * N + s]);
  }
}
__global__ void k_head_final(const float* __restrict__ hacc, const float* __restrict__ u_all,
                             const float* __restrict__ c1, const float* __restrict__ c2,
                             const float* __restrict__ cA, const float* __restrict__ bl1,
                             const float* __restrict__ bl2, const float* __restrict__ bl3,
                             float* __restrict__ out, int N) {
  int n = blockIdx.x * blockDim.x + threadIdx.x;
  if (n >= N) return;
  float bls[3] = {bl1[0], bl2[0], bl3[0]};
  float cnts[3] = {c1[n], c2[n], cA[n]};
  float s = 0.f;
#pragma unroll
  for (int i = 0; i < 9; ++i) {
    int g = i % 3;
    s += hacc[(size_t)i * N + n] / fmaxf(cnts[g], 1.f) + bls[g] + u_all[(size_t)i * N + n];
  }
  out[n] = 1.f / (1.f + expf(-s));
}

// ---------- host ----------
extern "C" void kernel_launch(void* const* d_in, const int* in_sizes, int n_in,
                              void* d_out, int out_size, void* d_ws, size_t ws_size,
                              hipStream_t stream) {
  (void)n_in; (void)out_size; (void)ws_size;
  const float* x    = (const float*)d_in[0];
  const int*   ei   = (const int*)d_in[1];
  const float* attr = (const float*)d_in[2];
  const int N = in_sizes[0] / 1028;
  const int E = in_sizes[2];
  const int* src = ei;
  const int* dst = ei + E;
  auto P = [&](int i) { return (const float*)d_in[3 + i]; };

  // workspace layout
  char* w = (char*)d_ws;
  size_t off = 0;
  auto alloc = [&](size_t b) { size_t p = off; off += (b + 255) & ~(size_t)255; return p; };
  size_t o_wcat = alloc((size_t)1024 * 64 * 2);
  int hids[5]  = {64, 64, 64, 128, 256};     // groups: ec11, ec12, ec13, ecO, ecO2
  int wbase[5] = {16, 28, 32, 20, 24};       // param index of <grp>_w1
  size_t o_w1p[5], o_w2p[5];
  for (int g = 0; g < 5; ++g) {
    o_w1p[g] = alloc((size_t)128 * hids[g] * 2);
    o_w2p[g] = alloc((size_t)hids[g] * 64 * 2);
  }
  size_t o_w4 = alloc(4 * 64 * 4), o_bf = alloc(64 * 4);
  size_t NE64 = (size_t)N * 64;
  size_t o_h0 = alloc(NE64 * 4), o_tmp = alloc(NE64 * 4), o_sums = alloc(NE64 * 4);
  size_t o_macc = alloc(3 * NE64 * 4);
  size_t o_xb = alloc(9 * NE64 * 4);
  size_t o_stats = alloc(128 * 4);
  size_t o_cnt = alloc((size_t)3 * N * 4);
  size_t o_t = alloc((size_t)9 * N * 4), o_u = alloc((size_t)9 * N * 4), o_hacc = alloc((size_t)9 * N * 4);

  float* h0   = (float*)(w + o_h0);
  float* tmp  = (float*)(w + o_tmp);
  float* sums = (float*)(w + o_sums);
  unsigned* macc = (unsigned*)(w + o_macc);
  float* xb   = (float*)(w + o_xb);
  float* stats = (float*)(w + o_stats);
  float* cnt  = (float*)(w + o_cnt);
  float* t_all = (float*)(w + o_t);
  float* u_all = (float*)(w + o_u);
  float* hacc = (float*)(w + o_hacc);
  const float invN = 1.0f / (float)N;

  // 1) pack weights to bf16 WMMA-B order
  auto pack = [&](const float* Wa, const float* Wb, int Ksplit, int K, int Nout, size_t oo) {
    int total = (K / 32) * (Nout / 16) * 32;
    k_pack_b<<<(total + 127) / 128, 128, 0, stream>>>(Wa, Wb, Ksplit, K, Nout,
                                                      (unsigned short*)(w + oo));
  };
  pack(P(4), P(2), 512, 1024, 64, o_wcat);                 // Wcat = [l012_w ; l011_w[:512]]
  for (int g = 0; g < 5; ++g) {
    pack(P(wbase[g] + 0), P(wbase[g] + 0), 128, 128, hids[g], o_w1p[g]);
    pack(P(wbase[g] + 2), P(wbase[g] + 2), hids[g], hids[g], 64, o_w2p[g]);
  }
  k_fold_spf<<<1, 64, 0, stream>>>(P(0), P(1), P(2), P(3), P(5), (float*)(w + o_w4), (float*)(w + o_bf));

  // 2) node projection GEMM + BN(bn01) + relu -> h0
  k_node_gemm<<<(N + 15) / 16, 32, 0, stream>>>(x, (const unsigned short*)(w + o_wcat),
                                                (const float*)(w + o_w4), (const float*)(w + o_bf), h0, N);
  k_zero_f32<<<1, 128, 0, stream>>>(stats, 128);
  k_bn_stats<<<1024, 256, 0, stream>>>(h0, N * 64, stats);
  k_bn_relu<<<(N * 64 + 255) / 256, 256, 0, stream>>>(h0, N * 64, stats, P(6), P(7), invN);

  // 3) neighbor counts per mask
  k_zero_f32<<<256, 256, 0, stream>>>(cnt, 3 * N);
  k_counts<<<(E + 255) / 256, 256, 0, stream>>>(dst, attr, E, cnt, cnt + N, cnt + 2 * N);

  // 4) conv groups -> branches
  // branch order: x1,x2,x3,x11,x22,x33,x111,x222,x333
  int g_nb[5]        = {1, 1, 1, 3, 3};
  int g_slot[5][3]   = {{0}, {0}, {0}, {0, 1, 2}, {0, 1, 2}};
  int g_cmask[5][3]  = {{0}, {1}, {2}, {0, 1, 2}, {0, 1, 2}};
  int g_bidx[5][3]   = {{0}, {1}, {2}, {3, 4, 5}, {6, 7, 8}};

  for (int g = 0; g < 5; ++g) {
    unsigned* aptr[3] = {nullptr, nullptr, nullptr};
    for (int b = 0; b < g_nb[g]; ++b) {
      unsigned* slotp = macc + (size_t)g_slot[g][b] * NE64;
      aptr[g_cmask[g][b]] = slotp;
      k_fill_u32<<<2048, 256, 0, stream>>>(slotp, N * 64, NEGINF_KEY);
    }
    k_edge_mlp<<<(E + EPB - 1) / EPB, 32, 0, stream>>>(
        h0, src, dst, attr,
        (const unsigned short*)(w + o_w1p[g]), P(wbase[g] + 1),
        (const unsigned short*)(w + o_w2p[g]), P(wbase[g] + 3),
        hids[g], E, aptr[0], aptr[1], aptr[2]);

    for (int b = 0; b < g_nb[g]; ++b) {
      int bi = g_bidx[g][b];
      unsigned* acc = macc + (size_t)g_slot[g][b] * NE64;
      k_seg_finalize<<<2048, 256, 0, stream>>>(acc, tmp, N * 64);
      k_zero_f32<<<1, 128, 0, stream>>>(stats, 128);
      k_bn_stats<<<1024, 256, 0, stream>>>(tmp, N * 64, stats);
      int bng = bi % 3;  // bn11 / bn12 / bn13
      k_bn_relu<<<(N * 64 + 255) / 256, 256, 0, stream>>>(tmp, N * 64, stats, P(8 + 2 * bng), P(9 + 2 * bng), invN);
      // SAGE (s21)
      int mm = (bi == 1) ? 1 : (bi == 2) ? 2 : 0;  // sage mask
      k_zero_f32<<<2048, 256, 0, stream>>>(sums, N * 64);
      k_sage_scatter<<<4096, 256, 0, stream>>>(tmp, src, dst, attr, mm, E, sums);
      float* xbI = xb + (size_t)bi * NE64;
      k_sage_linear<<<N, 64, 0, stream>>>(tmp, sums, cnt + (size_t)mm * N, P(36), P(37), P(38), xbI, N);
      // BN(bn21) + relu in place
      k_zero_f32<<<1, 128, 0, stream>>>(stats, 128);
      k_bn_stats<<<1024, 256, 0, stream>>>(xbI, N * 64, stats);
      k_bn_relu<<<(N * 64 + 255) / 256, 256, 0, stream>>>(xbI, N * 64, stats, P(14), P(15), invN);
    }
  }

  // 5) heads: per-branch dot with wl/wr, scatter mean-sums, final sigmoid
  k_zero_f32<<<512, 256, 0, stream>>>(hacc, 9 * N);
  for (int bi = 0; bi < 9; ++bi) {
    int hg = bi % 3;                // s31 / s32 / s33
    int base = 39 + hg * 3;         // wl, bl, wr
    k_head_dots<<<(N + 255) / 256, 256, 0, stream>>>(xb + (size_t)bi * NE64, P(base), P(base + 2),
                                                     t_all + (size_t)bi * N, u_all + (size_t)bi * N, N);
  }
  k_head_scatter<<<(E + 255) / 256, 256, 0, stream>>>(t_all, src, dst, attr, hacc, E, N);
  k_head_final<<<(N + 255) / 256, 256, 0, stream>>>(hacc, u_all, cnt, cnt + N, cnt + 2 * N,
                                                   P(40), P(43), P(46), (float*)d_out, N);
}
